// Readout_29575144800944
// MI455X (gfx1250) — matrix operand order repaired and checked by reading, not compiled, not run on hardware
//
#include <hip/hip_runtime.h>
#include <hip/hip_bf16.h>

#define N_MODULES 8
#define D_IN   4096
#define D_HID  4096
#define OUT_SZ 10
#define BATCH  8192
#define MOD_W  512   // per-module in/hid width

typedef __attribute__((ext_vector_type(16))) __bf16 v16bf_t;
typedef __attribute__((ext_vector_type(8)))  float  v8f_t;

union FragAB { v16bf_t v; unsigned u[8]; };

// round-to-nearest-even fp32 -> bf16 (low 16 bits of result)
__device__ __forceinline__ unsigned bfround(float f) {
    unsigned u = __float_as_uint(f);
    return (u + 0x7FFFu + ((u >> 16) & 1u)) >> 16;
}
__device__ __forceinline__ float bf2f(unsigned h) { return __uint_as_float(h << 16); }

// split two floats into packed bf16 hi pair + bf16 residual (lo) pair
__device__ __forceinline__ void split2(float f0, float f1, unsigned& hi, unsigned& lo) {
    unsigned h0 = bfround(f0), h1 = bfround(f1);
    unsigned l0 = bfround(f0 - bf2f(h0));
    unsigned l1 = bfround(f1 - bf2f(h1));
    hi = (h0 & 0xFFFFu) | (h1 << 16);
    lo = (l0 & 0xFFFFu) | (l1 << 16);
}

// ---- One-shot precompute: split diagonal W1 blocks into bf16 hi/lo planes ----
// Layout: plane[((m*512 + n) << 9) + k]  (K-contiguous per hidden unit n)
__global__ __launch_bounds__(256)
void w1_split_kernel(const float* __restrict__ W1,
                     unsigned short* __restrict__ whi,
                     unsigned short* __restrict__ wlo) {
    size_t e = ((size_t)blockIdx.x * 256 + threadIdx.x) * 8;   // 8 elements/thread
    int m   = (int)(e >> 18);            // 512*512 = 2^18 elements per module
    int rem = (int)(e & 0x3FFFFu);
    int n   = rem >> 9;
    int k   = rem & (MOD_W - 1);
    const float* src = W1 + (size_t)(m * MOD_W + n) * D_IN + m * MOD_W + k;
    float4 a = *(const float4*)(src);
    float4 b = *(const float4*)(src + 4);
    unsigned h[4], l[4];
    split2(a.x, a.y, h[0], l[0]);
    split2(a.z, a.w, h[1], l[1]);
    split2(b.x, b.y, h[2], l[2]);
    split2(b.z, b.w, h[3], l[3]);
    *(uint4*)(whi + e) = make_uint4(h[0], h[1], h[2], h[3]);
    *(uint4*)(wlo + e) = make_uint4(l[0], l[1], l[2], l[3]);
}

// ---- Fused 2-layer block-diagonal MLP ----
// PRE=true: B fragments loaded directly from precomputed bf16 planes (no VALU conversion)
template <bool PRE>
__global__ __launch_bounds__(256, 1)
void readout_fused(const float* __restrict__ x,
                   const float* __restrict__ W1,
                   const float* __restrict__ b1,
                   const float* __restrict__ W2,
                   const float* __restrict__ b2,
                   const unsigned short* __restrict__ whi,
                   const unsigned short* __restrict__ wlo,
                   float* __restrict__ out) {
    __shared__ unsigned xs_hi[2][64][16];  // double-buffered x chunk, bf16 pairs (hi)
    __shared__ unsigned xs_lo[2][64][16];  // residual (lo)
    __shared__ float    hs[64][68];        // h chunk staging (64x64, padded)
    __shared__ float    w2s[OUT_SZ][MOD_W];
    __shared__ float    b1s[MOD_W];

    const int t     = threadIdx.x;
    const int lane  = t & 31;
    const int wv    = t >> 5;            // wave id 0..7 -> 64-col group
    const int m     = blockIdx.y;        // module
    const int r0    = blockIdx.x * 64;   // batch row base
    const int mbase = m * MOD_W;

    const int nloc  = lane & 15;
    const int khalf = lane >> 4;
    const int kb    = khalf * 8;         // A-fragment pair base (dwords)

    for (int i = t; i < OUT_SZ * MOD_W; i += 256) {
        int o = i >> 9, n = i & (MOD_W - 1);
        w2s[o][n] = W2[(size_t)(m * OUT_SZ + o) * D_HID + mbase + n];
    }
    for (int i = t; i < MOD_W; i += 256) b1s[i] = b1[mbase + i];

    v8f_t zero;
#pragma unroll
    for (int i = 0; i < 8; ++i) zero[i] = 0.0f;
    v8f_t acc[4][4];
#pragma unroll
    for (int rt = 0; rt < 4; ++rt)
#pragma unroll
        for (int ct = 0; ct < 4; ++ct) acc[rt][ct] = zero;

    // ---------------- Layer 1: bf16x3 WMMA GEMM over K = 512 ----------------
    // Software-pipelined x staging: chunk (k0+32) is loaded into registers while
    // chunk k0 is being consumed; one barrier per K-step (buffer p is rewritten
    // only in iteration i+2, i.e. after barrier i+1 which implies all reads done).
    const int srow = t >> 2;             // staging row this thread fills
    const int c8   = (t & 3) * 8;        // 8 consecutive K elements
    const int p8   = c8 >> 1;            // pair index
    const float* xrow = x + (size_t)(r0 + srow) * D_IN + mbase + c8;

    float4 xa = *(const float4*)(xrow);
    float4 xb = *(const float4*)(xrow + 4);

    for (int k0 = 0; k0 < MOD_W; k0 += 32) {
        const int pb = (k0 >> 5) & 1;
        split2(xa.x, xa.y, xs_hi[pb][srow][p8 + 0], xs_lo[pb][srow][p8 + 0]);
        split2(xa.z, xa.w, xs_hi[pb][srow][p8 + 1], xs_lo[pb][srow][p8 + 1]);
        split2(xb.x, xb.y, xs_hi[pb][srow][p8 + 2], xs_lo[pb][srow][p8 + 2]);
        split2(xb.z, xb.w, xs_hi[pb][srow][p8 + 3], xs_lo[pb][srow][p8 + 3]);
        __syncthreads();

        // issue next chunk's global loads now; waited on only next iteration
        if (k0 + 32 < MOD_W) {
            xa = *(const float4*)(xrow + k0 + 32);
            xb = *(const float4*)(xrow + k0 + 36);
        }

        // A fragments (16x32 bf16), vectorized LDS reads (2 x b128 per frag)
        FragAB ah[4], al[4];
#pragma unroll
        for (int rt = 0; rt < 4; ++rt) {
            int mrow = rt * 16 + nloc;
            *(uint4*)&ah[rt].u[0] = *(const uint4*)&xs_hi[pb][mrow][kb];
            *(uint4*)&ah[rt].u[4] = *(const uint4*)&xs_hi[pb][mrow][kb + 4];
            *(uint4*)&al[rt].u[0] = *(const uint4*)&xs_lo[pb][mrow][kb];
            *(uint4*)&al[rt].u[4] = *(const uint4*)&xs_lo[pb][mrow][kb + 4];
        }

#pragma unroll
        for (int ct = 0; ct < 4; ++ct) {
            FragAB bhi, blo;
            if constexpr (PRE) {
                // direct bf16 plane loads: 16 contiguous bf16 = 2 x b128 per plane
                size_t off = ((size_t)(m * MOD_W + wv * 64 + ct * 16 + nloc) << 9)
                             + k0 + khalf * 16;
                const unsigned short* rh = whi + off;
                const unsigned short* rl = wlo + off;
                *(uint4*)&bhi.u[0] = *(const uint4*)(rh);
                *(uint4*)&bhi.u[4] = *(const uint4*)(rh + 8);
                *(uint4*)&blo.u[0] = *(const uint4*)(rl);
                *(uint4*)&blo.u[4] = *(const uint4*)(rl + 8);
            } else {
                const float* wrow = W1
                    + (size_t)(mbase + wv * 64 + ct * 16 + nloc) * D_IN
                    + mbase + k0 + khalf * 16;
                float4 f0 = *(const float4*)(wrow + 0);
                float4 f1 = *(const float4*)(wrow + 4);
                float4 f2 = *(const float4*)(wrow + 8);
                float4 f3 = *(const float4*)(wrow + 12);
                split2(f0.x, f0.y, bhi.u[0], blo.u[0]);
                split2(f0.z, f0.w, bhi.u[1], blo.u[1]);
                split2(f1.x, f1.y, bhi.u[2], blo.u[2]);
                split2(f1.z, f1.w, bhi.u[3], blo.u[3]);
                split2(f2.x, f2.y, bhi.u[4], blo.u[4]);
                split2(f2.z, f2.w, bhi.u[5], blo.u[5]);
                split2(f3.x, f3.y, bhi.u[6], blo.u[6]);
                split2(f3.z, f3.w, bhi.u[7], blo.u[7]);
            }

#pragma unroll
            for (int rt = 0; rt < 4; ++rt) {
                // hi*hi + hi*lo + lo*hi (drop lo*lo) -> ~fp32 accuracy, f32 accumulate
                acc[rt][ct] = __builtin_amdgcn_wmma_f32_16x16x32_bf16(
                    false, ah[rt].v, false, bhi.v, (short)0, acc[rt][ct], false, false);
                acc[rt][ct] = __builtin_amdgcn_wmma_f32_16x16x32_bf16(
                    false, ah[rt].v, false, blo.v, (short)0, acc[rt][ct], false, false);
                acc[rt][ct] = __builtin_amdgcn_wmma_f32_16x16x32_bf16(
                    false, al[rt].v, false, bhi.v, (short)0, acc[rt][ct], false, false);
            }
        }
    }

    // ---------------- Layer 2: relu+bias -> LDS chunks -> 512-dot with W2 ----------------
    float yacc[3] = {0.f, 0.f, 0.f};
    const int row_l = t & 63;
    const int g     = t >> 6;   // o in {g, g+4, g+8} (skip >= 10)

    for (int cw = 0; cw < 8; ++cw) {
        __syncthreads();
        if (wv == cw) {
            // C tile: row = rt*16 + 8*(lane>=16) + v ; col = ct*16 + (lane&15)
#pragma unroll
            for (int rt = 0; rt < 4; ++rt)
#pragma unroll
                for (int ct = 0; ct < 4; ++ct) {
                    int col = ct * 16 + nloc;
                    float bb = b1s[cw * 64 + col];
#pragma unroll
                    for (int v = 0; v < 8; ++v) {
                        float hval = acc[rt][ct][v] + bb;
                        hs[rt * 16 + khalf * 8 + v][col] = hval > 0.f ? hval : 0.f;
                    }
                }
        }
        __syncthreads();
        for (int n = 0; n < 64; ++n) {
            float hv = hs[row_l][n];
#pragma unroll
            for (int j = 0; j < 3; ++j) {
                int o = g + 4 * j;
                if (o < OUT_SZ) yacc[j] += hv * w2s[o][cw * 64 + n];
            }
        }
    }

#pragma unroll
    for (int j = 0; j < 3; ++j) {
        int o = g + 4 * j;
        if (o < OUT_SZ)
            out[(size_t)(r0 + row_l) * (N_MODULES * OUT_SZ) + m * OUT_SZ + o]
                = yacc[j] + b2[m * OUT_SZ + o];
    }
}

extern "C" void kernel_launch(void* const* d_in, const int* in_sizes, int n_in,
                              void* d_out, int out_size, void* d_ws, size_t ws_size,
                              hipStream_t stream) {
    const float* x  = (const float*)d_in[0];
    const float* W1 = (const float*)d_in[1];
    const float* b1 = (const float*)d_in[2];
    const float* W2 = (const float*)d_in[3];
    const float* b2 = (const float*)d_in[4];
    float* out = (float*)d_out;

    dim3 grid(BATCH / 64, N_MODULES);   // 128 row-tiles x 8 modules = 1024 WGs

    const size_t plane_elems = (size_t)N_MODULES * MOD_W * MOD_W;   // 2M bf16 per plane
    const size_t need = plane_elems * 2 * sizeof(unsigned short);   // hi + lo = 8 MB

    if (ws_size >= need) {
        unsigned short* whi = (unsigned short*)d_ws;
        unsigned short* wlo = whi + plane_elems;
        // 2M elements, 8 per thread, 256 threads/block -> 1024 blocks
        w1_split_kernel<<<dim3(1024), 256, 0, stream>>>(W1, whi, wlo);
        readout_fused<true><<<grid, 256, 0, stream>>>(x, W1, b1, W2, b2, whi, wlo, out);
    } else {
        readout_fused<false><<<grid, 256, 0, stream>>>(x, W1, b1, W2, b2,
                                                       nullptr, nullptr, out);
    }
}
